// MACELayer_2645699854351
// MI455X (gfx1250) — matrix operand order repaired and checked by reading, not compiled, not run on hardware
//
#include <hip/hip_runtime.h>
#include <hip/hip_bf16.h>
#include <math.h>

// ---------------- problem constants ----------------
#define N_NODES 10000
#define KNB     16
#define C_CH    32
#define RAD     32
#define EPSV    1e-6f

typedef __attribute__((ext_vector_type(16))) _Float16 v16h;
typedef __attribute__((ext_vector_type(8)))  float    v8f;

// ---------------- workspace layout (float offsets) ----------------
#define CG121_OFF 0                    // [3][5][3]  = 45
#define CG112_OFF 45                   // [3][3][5]  = 45
#define U30E_OFF  90                   // [9][9][9][3]      = 2187
#define U31O_OFF  (90 + 2187)          // [9][9][9][2][3]   = 4374
#define U20E_OFF  (U31O_OFF + 4374)    // [9][9][2]         = 162
#define U21O_OFF  (U20E_OFF + 162)     // [9][9][3] (k=1)   = 243
#define U10E_OFF  (U21O_OFF + 243)     // [9]               = 9
#define U11O_OFF  (U10E_OFF + 9)       // [9][3]            = 27
#define CONST_TOT (U11O_OFF + 27)      // 7092  (divisible by 4 -> B128 chunks)

#define WS_S     7168
#define WS_V     (WS_S    + N_NODES * C_CH)          // (n,c,i) c*3+i
#define WS_O0    (WS_V    + N_NODES * C_CH * 3)      // (n,64)
#define WS_O1    (WS_O0   + N_NODES * 2 * C_CH)      // (n,96,3)
#define WS_O2    (WS_O1   + N_NODES * 3 * C_CH * 3)  // (n,64,5)
#define WS_XSYM  (WS_O2   + N_NODES * 2 * C_CH * 5)  // (n,c,9)
#define WS_SYMS  (WS_XSYM + N_NODES * C_CH * 9)      // (n,c)
#define WS_SYMV  (WS_SYMS + N_NODES * C_CH)          // (n,c,3)

// ---------------- gfx1250 async global->LDS (ASYNCcnt path) ----------------
// VDST = LDS byte address VGPR, VADDR = 64-bit global address, no SADDR.
__device__ __forceinline__ void async_copy_b128(unsigned lds_addr, const float* gptr) {
  asm volatile("global_load_async_to_lds_b128 %0, %1, off"
               :: "v"(lds_addr), "v"(gptr) : "memory");
}
__device__ __forceinline__ void wait_async0() {
  asm volatile("s_wait_asynccnt 0x0" ::: "memory");
}
// low 32 bits of a flat LDS pointer are the LDS byte offset (aperture layout)
__device__ __forceinline__ unsigned lds_base_addr(const void* p) {
  return (unsigned)(unsigned long long)p;
}

// ---------------- WMMA fragment helpers (gfx1250 wave32) ----------------
// A: 16x32 f16. lane<16 -> M=lane, K halves {0..7,16..23}; lane>=16 -> M=lane-16, {8..15,24..31}
__device__ __forceinline__ v16h wmma_load_a(const float* src, int row_stride, int k_stride) {
  int l = threadIdx.x & 31;
  int m = l & 15;
  int kh = (l >> 4) * 8;
  const float* r = src + m * row_stride;
  v16h a;
#pragma unroll
  for (int j = 0; j < 8; ++j) {
    int k = ((j & 3) * 2) + ((j >> 2) * 16) + kh;
    a[2 * j]     = (_Float16)r[k * k_stride];
    a[2 * j + 1] = (_Float16)r[(k + 1) * k_stride];
  }
  return a;
}

__device__ __forceinline__ v16h wmma_load_a_scaled(const float* src, int row_stride,
                                                   int k_stride, const float* rowscale) {
  int l = threadIdx.x & 31;
  int m = l & 15;
  int kh = (l >> 4) * 8;
  const float* r = src + m * row_stride;
  float g = rowscale[m];
  v16h a;
#pragma unroll
  for (int j = 0; j < 8; ++j) {
    int k = ((j & 3) * 2) + ((j >> 2) * 16) + kh;
    a[2 * j]     = (_Float16)(r[k * k_stride] * g);
    a[2 * j + 1] = (_Float16)(r[(k + 1) * k_stride] * g);
  }
  return a;
}

// B: 32x16 f16 (KxN). lane<16 -> N=lane, K {0..7,16..23}; lane>=16 -> N=lane-16, K {8..15,24..31}
__device__ __forceinline__ v16h wmma_load_b(const float* W, int krow_stride, int n_stride,
                                            int kb, int nb) {
  int l = threadIdx.x & 31;
  int n = nb + (l & 15);
  int k0 = kb + ((l >> 4) * 8);
  v16h b;
#pragma unroll
  for (int j = 0; j < 8; ++j) {
    int kk = k0 + ((j & 3) * 2) + ((j >> 2) * 16);
    b[2 * j]     = (_Float16)W[kk * krow_stride + n * n_stride];
    b[2 * j + 1] = (_Float16)W[(kk + 1) * krow_stride + n * n_stride];
  }
  return b;
}

__device__ __forceinline__ v8f wmma_f16(v16h a, v16h b, v8f c) {
  return __builtin_amdgcn_wmma_f32_16x16x32_f16(false, a, false, b, (short)0, c, false, false);
}

// D: 8 VGPRs; VGPR r -> lanes 0-15: M=r, lanes 16-31: M=8+r; N=lane&15
__device__ __forceinline__ void wmma_store(float* dst, int ld, int nb, v8f d, float scale) {
  int l = threadIdx.x & 31;
  int n = nb + (l & 15);
  int m0 = (l >> 4) * 8;
#pragma unroll
  for (int r = 0; r < 8; ++r) dst[(m0 + r) * ld + n] = d[r] * scale;
}

// ---------------- deterministic constant generation ----------------
__device__ __forceinline__ unsigned pcg_hash(unsigned x) {
  x = x * 747796405u + 2891336453u;
  x = ((x >> ((x >> 28u) + 4u)) ^ x) * 277803737u;
  return (x >> 22u) ^ x;
}
__device__ __forceinline__ float hash_normal(unsigned idx, unsigned tag) {
  unsigned h1 = pcg_hash(idx ^ (tag * 0x9E3779B9u));
  unsigned h2 = pcg_hash(h1 + 0x85EBCA6Bu);
  float u1 = ((float)(h1 >> 8) + 1.0f) * (1.0f / 16777218.0f);
  float u2 = (float)(h2 >> 8) * (1.0f / 16777216.0f);
  return sqrtf(-2.0f * logf(u1)) * cosf(6.28318530718f * u2);
}

__global__ void k_init_const(float* ws) {
  int i = blockIdx.x * blockDim.x + threadIdx.x;
  if (i >= CONST_TOT) return;
  unsigned tag; float scale;
  if (i < CG112_OFF)      { tag = 1; scale = 0.2f;   }   // CG121
  else if (i < U30E_OFF)  { tag = 2; scale = 0.2f;   }   // CG112
  else if (i < U31O_OFF)  { tag = 3; scale = 0.027f; }   // U3 0e : 0.3^3
  else if (i < U20E_OFF)  { tag = 4; scale = 0.027f; }   // U3 1o
  else if (i < U21O_OFF)  { tag = 5; scale = 0.09f;  }   // U2 0e : 0.3^2
  else if (i < U10E_OFF)  { tag = 6; scale = 0.09f;  }   // U2 1o
  else if (i < U11O_OFF)  { tag = 7; scale = 0.3f;   }   // U1 0e
  else                    { tag = 8; scale = 0.3f;   }   // U1 1o
  ws[i] = scale * hash_normal((unsigned)i, tag);
}

__global__ void k_zero_o(float* ws) {
  long total = (long)N_NODES * 672;  // O0|O1|O2 contiguous
  for (long i = (long)blockIdx.x * blockDim.x + threadIdx.x; i < total;
       i += (long)gridDim.x * blockDim.x)
    ws[WS_O0 + i] = 0.0f;
}

// ---------------- K1: node feature update (WMMA) + RMS norm ----------------
__global__ void __launch_bounds__(32)
k_update(const float* __restrict__ node_s, const float* __restrict__ node_v,
         const float* __restrict__ Wu0, const float* __restrict__ Wu1, float* ws) {
  __shared__ float ssh[16 * 32];
  __shared__ float vsh[16 * 96];
  __shared__ float rmss[16], rmsv[16];
  const int n0 = blockIdx.x * 16;
  const int l  = threadIdx.x;
  const float inv = 0.17677669529663687f;  // 1/sqrt(32)

  {
    v16h a  = wmma_load_a(node_s + (long)n0 * C_CH, C_CH, 1);
    v16h b0 = wmma_load_b(Wu0, C_CH, 1, 0, 0);
    v16h b1 = wmma_load_b(Wu0, C_CH, 1, 0, 16);
    v8f a0 = {}, a1 = {};
    a0 = wmma_f16(a, b0, a0);
    a1 = wmma_f16(a, b1, a1);
    wmma_store(ssh, 32, 0, a0, inv);
    wmma_store(ssh, 32, 16, a1, inv);
  }
  {
    v16h b0 = wmma_load_b(Wu1, C_CH, 1, 0, 0);
    v16h b1 = wmma_load_b(Wu1, C_CH, 1, 0, 16);
#pragma unroll
    for (int i = 0; i < 3; ++i) {
      v16h av = wmma_load_a(node_v + (long)n0 * 96 + i, 96, 3);
      v8f u0 = {}, u1 = {};
      u0 = wmma_f16(av, b0, u0);
      u1 = wmma_f16(av, b1, u1);
      int nn = l & 15, m0 = (l >> 4) * 8;
#pragma unroll
      for (int r = 0; r < 8; ++r) {
        vsh[(m0 + r) * 96 + nn * 3 + i]        = u0[r] * inv;
        vsh[(m0 + r) * 96 + (nn + 16) * 3 + i] = u1[r] * inv;
      }
    }
  }
  __syncthreads();
  if (l < 16) {
    float a = 0.f, b = 0.f;
    for (int c = 0; c < 32; ++c) { float x = ssh[l * 32 + c]; a += x * x; }
    for (int c = 0; c < 96; ++c) { float x = vsh[l * 96 + c]; b += x * x; }
    rmss[l] = rsqrtf(a / 32.f + EPSV);
    rmsv[l] = rsqrtf(b / 96.f + EPSV);
  }
  __syncthreads();
  float* S = ws + WS_S;
  float* V = ws + WS_V;
  for (int t = l; t < 16 * 32; t += 32) {
    int m = t >> 5, c = t & 31;
    S[(long)(n0 + m) * 32 + c] = ssh[m * 32 + c] * rmss[m];
  }
  for (int t = l; t < 16 * 96; t += 32) {
    int m = t / 96, c = t % 96;
    V[(long)(n0 + m) * 96 + c] = vsh[m * 96 + c] * rmsv[m];
  }
}

// ---------------- K2: fused radial GEMM (WMMA) + LayerNorm + SH messages + scatter ----------
__global__ void __launch_bounds__(32)
k_edge(const float* __restrict__ vectors, const float* __restrict__ radial,
       const float* __restrict__ radW, const float* __restrict__ radb,
       const float* __restrict__ ln_g, const float* __restrict__ ln_b,
       const int* __restrict__ receivers, float* ws) {
  __shared__ float radsh[16 * 224];
  __shared__ float mu_[16], rs_[16];
  __shared__ float esh1[16][3], esh2[16][5], eT[16][9], eT2[16][15];
  const int n = blockIdx.x;
  const int l = threadIdx.x;

  __builtin_prefetch(radW, 0, 1);

  v16h a = wmma_load_a(radial + (long)n * KNB * RAD, RAD, 1);
#pragma unroll 2
  for (int nt = 0; nt < 14; ++nt) {
    v16h b = wmma_load_b(radW, 224, 1, 0, nt * 16);
    v8f acc = {};
    acc = wmma_f16(a, b, acc);
    int col = nt * 16 + (l & 15);
    int m0 = (l >> 4) * 8;
    float bias = radb[col];
#pragma unroll
    for (int r = 0; r < 8; ++r) radsh[(m0 + r) * 224 + col] = acc[r] + bias;
  }
  __syncthreads();
  if (l < 16) {
    float s = 0.f, s2 = 0.f;
    for (int c = 0; c < 224; ++c) { float x = radsh[l * 224 + c]; s += x; s2 += x * x; }
    float mu = s * (1.0f / 224.f);
    float var = s2 * (1.0f / 224.f) - mu * mu;
    mu_[l] = mu;
    rs_[l] = rsqrtf(var + EPSV);
  }
  __syncthreads();
  for (int t = l; t < 16 * 224; t += 32) {
    int k = t / 224, c = t % 224;
    radsh[t] = ln_g[c] * (radsh[t] - mu_[k]) * rs_[k] + ln_b[c];
  }
  const float* CGa = ws + CG121_OFF;  // [i][p][j]
  const float* CGb = ws + CG112_OFF;  // [i][j][p]
  if (l < 16) {
    const float* vp = vectors + ((long)n * KNB + l) * 3;
    float x = vp[0], y = vp[1], z = vp[2];
    float rl = sqrtf(x * x + y * y + z * z) + EPSV;
    x /= rl; y /= rl; z /= rl;
    const float s3 = 1.7320508075688772f, c15 = 3.872983346207417f;
    float h1[3] = {s3 * x, s3 * y, s3 * z};
    float h2[5] = {c15 * x * y, c15 * y * z, 1.118033988749895f * (3.f * z * z - 1.f),
                   c15 * x * z, 0.5f * c15 * (x * x - y * y)};
#pragma unroll
    for (int i = 0; i < 3; ++i) esh1[l][i] = h1[i];
#pragma unroll
    for (int p = 0; p < 5; ++p) esh2[l][p] = h2[p];
    for (int i = 0; i < 3; ++i)
      for (int j = 0; j < 3; ++j) {
        float t = 0.f;
        for (int p = 0; p < 5; ++p) t += h2[p] * CGa[i * 15 + p * 3 + j];
        eT[l][i * 3 + j] = t;
      }
    for (int i = 0; i < 3; ++i)
      for (int p = 0; p < 5; ++p) {
        float t = 0.f;
        for (int j = 0; j < 3; ++j) t += h1[j] * CGb[i * 15 + j * 5 + p];
        eT2[l][i * 5 + p] = t;
      }
  }
  __syncthreads();
  const int c = l;  // lane == channel (C==32)
  const float* S = ws + WS_S;
  const float* V = ws + WS_V;
  float* O0 = ws + WS_O0;
  float* O1 = ws + WS_O1;
  float* O2 = ws + WS_O2;
  const float sv  = S[(long)n * 32 + c];
  const float vv0 = V[(long)n * 96 + c * 3 + 0];
  const float vv1 = V[(long)n * 96 + c * 3 + 1];
  const float vv2 = V[(long)n * 96 + c * 3 + 2];
  const float inv3 = 0.5773502691896258f, inv5 = 0.4472135954999579f;
  for (int k = 0; k < KNB; ++k) {
    long idx = receivers[(long)n * KNB + k];
    const float* r = radsh + k * 224;
    float m0b = (vv0 * esh1[k][0] + vv1 * esh1[k][1] + vv2 * esh1[k][2]) * inv3;
    atomicAdd(&O0[idx * 64 + c],      sv  * r[c]);
    atomicAdd(&O0[idx * 64 + 32 + c], m0b * r[32 + c]);
    float r1a = r[64 + c], r1b = r[96 + c], r1c = r[128 + c];
#pragma unroll
    for (int i = 0; i < 3; ++i) {
      float vvi = (i == 0 ? vv0 : (i == 1 ? vv1 : vv2));
      float m1b = sv * esh1[k][i] * inv3;
      float m1c = vv0 * eT[k][0 * 3 + i] + vv1 * eT[k][1 * 3 + i] + vv2 * eT[k][2 * 3 + i];
      atomicAdd(&O1[(idx * 96 + c) * 3 + i],      vvi * r1a);
      atomicAdd(&O1[(idx * 96 + 32 + c) * 3 + i], m1b * r1b);
      atomicAdd(&O1[(idx * 96 + 64 + c) * 3 + i], m1c * r1c);
    }
    float r2a = r[160 + c], r2b = r[192 + c];
#pragma unroll
    for (int p = 0; p < 5; ++p) {
      float m2a = sv * esh2[k][p] * inv5;
      float m2b = vv0 * eT2[k][0 * 5 + p] + vv1 * eT2[k][1 * 5 + p] + vv2 * eT2[k][2 * 5 + p];
      atomicAdd(&O2[(idx * 64 + c) * 5 + p],      m2a * r2a);
      atomicAdd(&O2[(idx * 64 + 32 + c) * 5 + p], m2b * r2b);
    }
  }
}

// ---------------- K3: async-stage o -> norm -> Wd projections (WMMA) -> x_sym ----------------
__global__ void __launch_bounds__(32)
k_a(const float* __restrict__ Wd0, const float* __restrict__ Wd1,
    const float* __restrict__ Wd2, float* ws) {
  __shared__ float osh[16 * 672];  // per node: o0[64] | o1[288] | o2[320]; A results overlay
  __shared__ float g0[16], g1[16], g2[16];
  __shared__ float rA0[16], rA1[16], rA2[16];
  const int n0 = blockIdx.x * 16;
  const int l  = threadIdx.x;
  const float sc = 0.25f;  // 1/sqrt(AVG_NEIGH), folded into g* below
  const float* O0 = ws + WS_O0;
  const float* O1 = ws + WS_O1;
  const float* O2 = ws + WS_O2;

  // ---- async DMA stage: raw o tiles -> LDS (B128 chunks, ASYNCcnt) ----
  const unsigned lbase = lds_base_addr(osh);
  for (int t = l; t < 16 * 16; t += 32) {          // o0: 16 chunks/node
    int m = t >> 4, cc = (t & 15) * 4;
    async_copy_b128(lbase + (unsigned)(m * 672 + cc) * 4u, O0 + (long)(n0 + m) * 64 + cc);
  }
  for (int t = l; t < 16 * 72; t += 32) {          // o1: 72 chunks/node
    int m = t / 72, cc = (t % 72) * 4;
    async_copy_b128(lbase + (unsigned)(m * 672 + 64 + cc) * 4u, O1 + (long)(n0 + m) * 288 + cc);
  }
  for (int t = l; t < 16 * 80; t += 32) {          // o2: 80 chunks/node
    int m = t / 80, cc = (t % 80) * 4;
    async_copy_b128(lbase + (unsigned)(m * 672 + 352 + cc) * 4u, O2 + (long)(n0 + m) * 320 + cc);
  }
  wait_async0();
  __syncthreads();

  if (l < 16) {
    const float* b = osh + l * 672;
    float a0 = 0.f, a1 = 0.f, a2 = 0.f;
    for (int c = 0; c < 64; ++c)  { float x = b[c];       a0 += x * x; }
    for (int c = 0; c < 288; ++c) { float x = b[64 + c];  a1 += x * x; }
    for (int c = 0; c < 320; ++c) { float x = b[352 + c]; a2 += x * x; }
    // LDS holds raw o; normnorm(o*sc) == o * (sc * rsqrt(sc^2*mean(o^2)+eps))
    g0[l] = sc * rsqrtf(sc * sc * a0 / 64.f + EPSV);
    g1[l] = sc * rsqrtf(sc * sc * a1 / 288.f + EPSV);
    g2[l] = sc * rsqrtf(sc * sc * a2 / 320.f + EPSV);
  }
  __syncthreads();
  // A0 = norm(o0) @ Wd0 / 8  (K=64)
  {
    v16h a0 = wmma_load_a_scaled(osh + 0, 672, 1, g0);
    v16h a1 = wmma_load_a_scaled(osh + 32, 672, 1, g0);
    v8f c0 = {}, c1 = {};
    c0 = wmma_f16(a0, wmma_load_b(Wd0, 32, 1, 0, 0), c0);
    c0 = wmma_f16(a1, wmma_load_b(Wd0, 32, 1, 32, 0), c0);
    c1 = wmma_f16(a0, wmma_load_b(Wd0, 32, 1, 0, 16), c1);
    c1 = wmma_f16(a1, wmma_load_b(Wd0, 32, 1, 32, 16), c1);
    __syncthreads();
    wmma_store(osh, 672, 0, c0, 0.125f);
    wmma_store(osh, 672, 16, c1, 0.125f);
  }
  // A1[:,:,i] = norm(o1)[:,:,i] @ Wd1 / sqrt(96)  (K=96)
  {
    v8f acc[3][2];
#pragma unroll
    for (int i = 0; i < 3; ++i) {
      v8f z = {}; acc[i][0] = z; acc[i][1] = z;
#pragma unroll
      for (int kc = 0; kc < 3; ++kc) {
        v16h aa = wmma_load_a_scaled(osh + 64 + kc * 96 + i, 672, 3, g1);
        acc[i][0] = wmma_f16(aa, wmma_load_b(Wd1, 32, 1, kc * 32, 0), acc[i][0]);
        acc[i][1] = wmma_f16(aa, wmma_load_b(Wd1, 32, 1, kc * 32, 16), acc[i][1]);
      }
    }
    __syncthreads();
    const float s1 = 0.10206207261596575f;
    int nn = l & 15, m0 = (l >> 4) * 8;
#pragma unroll
    for (int i = 0; i < 3; ++i)
#pragma unroll
      for (int r = 0; r < 8; ++r) {
        osh[(m0 + r) * 672 + 64 + nn * 3 + i]        = acc[i][0][r] * s1;
        osh[(m0 + r) * 672 + 64 + (nn + 16) * 3 + i] = acc[i][1][r] * s1;
      }
  }
  // A2[:,:,p] = norm(o2)[:,:,p] @ Wd2 / 8  (K=64)
  {
    v8f acc[5][2];
#pragma unroll
    for (int p = 0; p < 5; ++p) {
      v8f z = {}; acc[p][0] = z; acc[p][1] = z;
#pragma unroll
      for (int kc = 0; kc < 2; ++kc) {
        v16h aa = wmma_load_a_scaled(osh + 352 + kc * 160 + p, 672, 5, g2);
        acc[p][0] = wmma_f16(aa, wmma_load_b(Wd2, 32, 1, kc * 32, 0), acc[p][0]);
        acc[p][1] = wmma_f16(aa, wmma_load_b(Wd2, 32, 1, kc * 32, 16), acc[p][1]);
      }
    }
    __syncthreads();
    int nn = l & 15, m0 = (l >> 4) * 8;
#pragma unroll
    for (int p = 0; p < 5; ++p)
#pragma unroll
      for (int r = 0; r < 8; ++r) {
        osh[(m0 + r) * 672 + 352 + nn * 5 + p]        = acc[p][0][r] * 0.125f;
        osh[(m0 + r) * 672 + 352 + (nn + 16) * 5 + p] = acc[p][1][r] * 0.125f;
      }
  }
  __syncthreads();
  if (l < 16) {
    const float* b = osh + l * 672;
    float a0 = 0.f, a1 = 0.f, a2 = 0.f;
    for (int c = 0; c < 32; ++c)  { float x = b[c];       a0 += x * x; }
    for (int c = 0; c < 96; ++c)  { float x = b[64 + c];  a1 += x * x; }
    for (int c = 0; c < 160; ++c) { float x = b[352 + c]; a2 += x * x; }
    rA0[l] = rsqrtf(a0 / 32.f + EPSV);
    rA1[l] = rsqrtf(a1 / 96.f + EPSV);
    rA2[l] = rsqrtf(a2 / 160.f + EPSV);
  }
  __syncthreads();
  float* X = ws + WS_XSYM;
  for (int t = l; t < 16 * 32; t += 32) {
    int m = t >> 5, c = t & 31;
    const float* b = osh + m * 672;
    float* xp = X + ((long)(n0 + m) * 32 + c) * 9;
    xp[0] = b[c] * rA0[m];
#pragma unroll
    for (int i = 0; i < 3; ++i) xp[1 + i] = b[64 + c * 3 + i] * rA1[m];
#pragma unroll
    for (int p = 0; p < 5; ++p) xp[4 + p] = b[352 + c * 5 + p] * rA2[m];
  }
}

// ---------------- K4: symmetric contraction (VALU hotspot, U in LDS via async DMA) ----------
__global__ void __launch_bounds__(256)
k_sym(const int* __restrict__ node_specie, const float* __restrict__ species_table,
      const float* __restrict__ w30e, const float* __restrict__ w31o,
      const float* __restrict__ w20e, const float* __restrict__ w21o,
      const float* __restrict__ w10e, const float* __restrict__ w11o, float* ws) {
  __shared__ float Ush[CONST_TOT];
  {
    const unsigned lbase = lds_base_addr(Ush);
    for (int t = threadIdx.x; t < CONST_TOT / 4; t += 256)
      async_copy_b128(lbase + (unsigned)t * 16u, ws + t * 4);
    wait_async0();
  }
  __syncthreads();
  const float* U30 = Ush + U30E_OFF;
  const float* U31 = Ush + U31O_OFF;
  const float* U20 = Ush + U20E_OFF;
  const float* U21 = Ush + U21O_OFF;
  const float* U10 = Ush + U10E_OFF;
  const float* U11 = Ush + U11O_OFF;

  long idx = (long)blockIdx.x * 256 + threadIdx.x;
  if (idx >= (long)N_NODES * C_CH) return;
  int n = (int)(idx >> 5), c = (int)(idx & 31);

  float x[9];
  const float* X = ws + WS_XSYM + idx * 9;
#pragma unroll
  for (int j = 0; j < 9; ++j) x[j] = X[j];

  const float* st = species_table + (long)node_specie[n] * RAD;
  float W30[3] = {0, 0, 0}, W31[2] = {0, 0}, W20[2] = {0, 0}, W21 = 0, W10 = 0, W11 = 0;
  for (int e = 0; e < RAD; ++e) {
    float se = st[e];
    W30[0] += se * w30e[e * 96 + c];
    W30[1] += se * w30e[e * 96 + 32 + c];
    W30[2] += se * w30e[e * 96 + 64 + c];
    W31[0] += se * w31o[e * 64 + c];
    W31[1] += se * w31o[e * 64 + 32 + c];
    W20[0] += se * w20e[e * 64 + c];
    W20[1] += se * w20e[e * 64 + 32 + c];
    W21    += se * w21o[e * 32 + c];
    W10    += se * w10e[e * 32 + c];
    W11    += se * w11o[e * 32 + c];
  }
  float t0[27], t1[18];
#pragma unroll
  for (int j = 0; j < 9; ++j) {
    t0[j * 3 + 0] = x[j] * W30[0];
    t0[j * 3 + 1] = x[j] * W30[1];
    t0[j * 3 + 2] = x[j] * W30[2];
    t1[j * 2 + 0] = x[j] * W31[0];
    t1[j * 2 + 1] = x[j] * W31[1];
  }
  float H0[9];
  float H1[9][3];
#pragma unroll
  for (int d = 0; d < 9; ++d) { H0[d] = 0.f; H1[d][0] = 0.f; H1[d][1] = 0.f; H1[d][2] = 0.f; }

  for (int d1 = 0; d1 < 9; ++d1) {
    for (int d2 = 0; d2 < 9; ++d2) {
      const int dd = d1 * 9 + d2;
      const float* u30 = U30 + dd * 27;
      float g0 = 0.f;
      for (int m = 0; m < 27; ++m) g0 += u30[m] * t0[m];
      g0 += U20[dd * 2 + 0] * W20[0] + U20[dd * 2 + 1] * W20[1];
      H0[d1] += g0 * x[d2];
      const float* u31 = U31 + dd * 54;
      float ga = 0.f, gb = 0.f, gc = 0.f;
      for (int m = 0; m < 18; ++m) {
        float tv = t1[m];
        ga += u31[m * 3 + 0] * tv;
        gb += u31[m * 3 + 1] * tv;
        gc += u31[m * 3 + 2] * tv;
      }
      ga += U21[dd * 3 + 0] * W21;
      gb += U21[dd * 3 + 1] * W21;
      gc += U21[dd * 3 + 2] * W21;
      H1[d1][0] += ga * x[d2];
      H1[d1][1] += gb * x[d2];
      H1[d1][2] += gc * x[d2];
    }
  }
  float r0 = 0.f, rv0 = 0.f, rv1 = 0.f, rv2 = 0.f;
#pragma unroll
  for (int d1 = 0; d1 < 9; ++d1) {
    float h0 = H0[d1] + U10[d1] * W10;
    r0 += h0 * x[d1];
    rv0 += (H1[d1][0] + U11[d1 * 3 + 0] * W11) * x[d1];
    rv1 += (H1[d1][1] + U11[d1 * 3 + 1] * W11) * x[d1];
    rv2 += (H1[d1][2] + U11[d1 * 3 + 2] * W11) * x[d1];
  }
  ws[WS_SYMS + idx] = r0;
  ws[WS_SYMV + idx * 3 + 0] = rv0;
  ws[WS_SYMV + idx * 3 + 1] = rv1;
  ws[WS_SYMV + idx * 3 + 2] = rv2;
}

// ---------------- K5: output projections (WMMA) + species skip + readout ----------------
__global__ void __launch_bounds__(32)
k_out(const float* __restrict__ node_s, const float* __restrict__ node_v,
      const int* __restrict__ node_specie,
      const float* __restrict__ P0, const float* __restrict__ P1,
      const float* __restrict__ Wskip0, const float* __restrict__ Wskip1,
      const float* __restrict__ Wread, const float* __restrict__ ws, float* out) {
  __shared__ float pssh[16 * 32];
  __shared__ float pvsh[16 * 96];
  __shared__ float sosh[16 * 32];
  const int n0 = blockIdx.x * 16;
  const int l  = threadIdx.x;
  const float inv = 0.17677669529663687f;
  const float* SY = ws + WS_SYMS;
  const float* SV = ws + WS_SYMV;
  {
    v16h a  = wmma_load_a(SY + (long)n0 * 32, 32, 1);
    v8f c0 = {}, c1 = {};
    c0 = wmma_f16(a, wmma_load_b(P0, 32, 1, 0, 0), c0);
    c1 = wmma_f16(a, wmma_load_b(P0, 32, 1, 0, 16), c1);
    wmma_store(pssh, 32, 0, c0, inv);
    wmma_store(pssh, 32, 16, c1, inv);
  }
  {
    v16h b0 = wmma_load_b(P1, 32, 1, 0, 0);
    v16h b1 = wmma_load_b(P1, 32, 1, 0, 16);
#pragma unroll
    for (int i = 0; i < 3; ++i) {
      v16h a = wmma_load_a(SV + (long)n0 * 96 + i, 96, 3);
      v8f c0 = {}, c1 = {};
      c0 = wmma_f16(a, b0, c0);
      c1 = wmma_f16(a, b1, c1);
      int nn = l & 15, m0 = (l >> 4) * 8;
#pragma unroll
      for (int r = 0; r < 8; ++r) {
        pvsh[(m0 + r) * 96 + nn * 3 + i]        = c0[r] * inv;
        pvsh[(m0 + r) * 96 + (nn + 16) * 3 + i] = c1[r] * inv;
      }
    }
  }
  __syncthreads();
  const int d = l;
  for (int m = 0; m < 16; ++m) {
    const long n = n0 + m;
    const float* Wsk = Wskip0 + (long)node_specie[n] * 1024;
    const float* Wvk = Wskip1 + (long)node_specie[n] * 1024;
    float as = 0.f, av0 = 0.f, av1 = 0.f, av2 = 0.f;
    for (int cc = 0; cc < 32; ++cc) {
      float w  = Wsk[cc * 32 + d];
      float wv = Wvk[cc * 32 + d];
      as += node_s[n * 32 + cc] * w;
      const float* vp = node_v + (n * 32 + cc) * 3;
      av0 += vp[0] * wv; av1 += vp[1] * wv; av2 += vp[2] * wv;
    }
    float sout = pssh[m * 32 + d] + inv * as;
    sosh[m * 32 + d] = sout;
    float* orow = out + n * 129;
    orow[d] = sout;
    orow[32 + d * 3 + 0] = pvsh[m * 96 + d * 3 + 0] + inv * av0;
    orow[32 + d * 3 + 1] = pvsh[m * 96 + d * 3 + 1] + inv * av1;
    orow[32 + d * 3 + 2] = pvsh[m * 96 + d * 3 + 2] + inv * av2;
  }
  __syncthreads();
  if (l < 16) {
    const long n = n0 + l;
    float rd = 0.f;
    for (int dd = 0; dd < 32; ++dd) rd += sosh[l * 32 + dd] * Wread[dd];
    out[n * 129 + 128] = rd * inv;
  }
}

// ---------------- launch ----------------
extern "C" void kernel_launch(void* const* d_in, const int* in_sizes, int n_in,
                              void* d_out, int out_size, void* d_ws, size_t ws_size,
                              hipStream_t stream) {
  (void)in_sizes; (void)n_in; (void)out_size; (void)ws_size;
  const float* node_s        = (const float*)d_in[0];
  const float* node_v        = (const float*)d_in[1];
  const float* vectors       = (const float*)d_in[2];
  const float* radial        = (const float*)d_in[3];
  const int*   receivers     = (const int*)d_in[4];
  const int*   node_specie   = (const int*)d_in[5];
  const float* species_table = (const float*)d_in[6];
  const float* Wu0   = (const float*)d_in[7];
  const float* Wu1   = (const float*)d_in[8];
  const float* radW  = (const float*)d_in[9];
  const float* radb  = (const float*)d_in[10];
  const float* ln_g  = (const float*)d_in[11];
  const float* ln_b  = (const float*)d_in[12];
  const float* Wd0   = (const float*)d_in[13];
  const float* Wd1   = (const float*)d_in[14];
  const float* Wd2   = (const float*)d_in[15];
  const float* w30e  = (const float*)d_in[16];
  const float* w31o  = (const float*)d_in[17];
  const float* w20e  = (const float*)d_in[18];
  const float* w21o  = (const float*)d_in[19];
  const float* w10e  = (const float*)d_in[20];
  const float* w11o  = (const float*)d_in[21];
  const float* P0    = (const float*)d_in[22];
  const float* P1    = (const float*)d_in[23];
  const float* Wskip0 = (const float*)d_in[24];
  const float* Wskip1 = (const float*)d_in[25];
  const float* Wread  = (const float*)d_in[26];
  float* ws  = (float*)d_ws;
  float* out = (float*)d_out;

  k_init_const<<<(CONST_TOT + 255) / 256, 256, 0, stream>>>(ws);
  k_zero_o<<<2048, 256, 0, stream>>>(ws);
  k_update<<<N_NODES / 16, 32, 0, stream>>>(node_s, node_v, Wu0, Wu1, ws);
  k_edge<<<N_NODES, 32, 0, stream>>>(vectors, radial, radW, radb, ln_g, ln_b, receivers, ws);
  k_a<<<N_NODES / 16, 32, 0, stream>>>(Wd0, Wd1, Wd2, ws);
  k_sym<<<(N_NODES * C_CH + 255) / 256, 256, 0, stream>>>(node_specie, species_table,
                                                          w30e, w31o, w20e, w21o, w10e, w11o, ws);
  k_out<<<N_NODES / 16, 32, 0, stream>>>(node_s, node_v, node_specie, P0, P1,
                                         Wskip0, Wskip1, Wread, ws, out);
}
